// PatchMatch_2791728742565
// MI455X (gfx1250) — matrix operand order repaired and checked by reading, not compiled, not run on hardware
//
#include <hip/hip_runtime.h>
#include <hip/hip_bf16.h>

// ---------------------------------------------------------------------------
// PatchMatch NNF on MI455X (gfx1250, wave32, WMMA)
//
// Q = source patch features (16384 x 288), P = target patch features.
// dist[i,j] = Rp[i] - 2*(q_i . p_j) + Rq[j]  (faithful to reference's
// broadcast orientation: Rq indexed by COLUMN j, Rp by ROW i).
// argmin/min over j, fused into the GEMM.
//
// bf16 hi/lo split (q=qh+ql, p=ph+pl; q.p ~= qh.ph + ql.ph + qh.pl) gives
// near-fp32 dot products at bf16 WMMA throughput.
// ---------------------------------------------------------------------------

typedef __attribute__((ext_vector_type(16))) __bf16 bf16x16;
typedef __attribute__((ext_vector_type(8)))  float  float8;

#define NPTS   16384          // 128*128 points per map
#define NTILE  1024           // NPTS / 16
#define KSTEPS 9              // 288 / 32
#define BLK    512            // 16x32 bf16 elements per WMMA fragment block
#define MATSZ  ((size_t)NTILE * KSTEPS * BLK)   // 4,718,592 bf16 per matrix
#define TILE_BYTES (KSTEPS * BLK * 2)           // 9216 B per (hi or lo) tile
#define TILE_CHUNKS (2 * TILE_BYTES / 16)       // 1152 x 16B per hi+lo tile

// Scheduling-group pins: 0x100 = DS read, 0x008 = MFMA/WMMA.
#if defined(__has_builtin)
#if __has_builtin(__builtin_amdgcn_sched_group_barrier)
#define SGB(mask, cnt) __builtin_amdgcn_sched_group_barrier((mask), (cnt), 0)
#endif
#endif
#ifndef SGB
#define SGB(mask, cnt)
#endif

__device__ __forceinline__ int clamp127(int v) {
    return v < 0 ? 0 : (v > 127 ? 127 : v);
}

// ---------------------------------------------------------------------------
// Stage 1: build WMMA-native swizzled patch matrices (hi + lo bf16 split).
//   blockIdx.y == 0 : source map -> Qh/Ql in A-matrix (16x32) lane layout
//   blockIdx.y == 1 : target map -> Ph/Pl in B-matrix (32x16) lane layout
// K ordering: k = 32*shift + channel (dot product is K-permutation invariant).
// ---------------------------------------------------------------------------
__global__ __launch_bounds__(256) void build_patches_kernel(
    const float* __restrict__ src, const float* __restrict__ tgt,
    __bf16* __restrict__ Qh, __bf16* __restrict__ Ql,
    __bf16* __restrict__ Ph, __bf16* __restrict__ Pl)
{
    const int tid  = blockIdx.x * 256 + threadIdx.x;   // 0 .. 294911
    const int lane = tid & 31;
    const int blk  = tid >> 5;                         // 0 .. 9215
    if (blk >= NTILE * KSTEPS) return;
    const int t = blk / KSTEPS;                        // row/col tile
    const int s = blk % KSTEPS;                        // patch shift 0..8
    const bool isB = (blockIdx.y == 1);
    const float* map = isB ? tgt : src;

    const int dy = s / 3 - 1, dx = s % 3 - 1;
    const int pt = t * 16 + (lane & 15);               // point index (M or N)
    const int sy = clamp127((pt >> 7) + dy);
    const int sx = clamp127((pt & 127) + dx);
    const int off = sy * 128 + sx;

    union { bf16x16 v; __bf16 e[16]; } hi, lo;
    const int upper = (lane >= 16);
#pragma unroll
    for (int v = 0; v < 8; ++v) {
#pragma unroll
        for (int h = 0; h < 2; ++h) {
            int k;
            if (!isB) {
                // A-matrix 16x32 bf16 layout (ISA 7.12.2):
                // lanes 0-15 hold K={0..7,16..23}, lanes 16-31 K={8..15,24..31}
                const int hi16 = v >> 2;
                const int base = v & 3;
                k = hi16 * 16 + base * 2 + h + (upper ? 8 : 0);
            } else {
                // B-matrix 32x16 bf16 layout (same K order as A):
                // lanes 0-15 hold K={0..7,16..23}, lanes 16-31 K={8..15,24..31}
                k = (v >> 2) * 16 + (v & 3) * 2 + h + (upper ? 8 : 0);
            }
            const float val = map[k * NPTS + off];     // k == channel here
            const __bf16 hh = (__bf16)val;
            hi.e[v * 2 + h] = hh;
            lo.e[v * 2 + h] = (__bf16)(val - (float)hh);
        }
    }
    const size_t dst = (size_t)blk * BLK + (size_t)lane * 16;
    if (!isB) {
        *(bf16x16*)(Qh + dst) = hi.v;
        *(bf16x16*)(Ql + dst) = lo.v;
    } else {
        *(bf16x16*)(Ph + dst) = hi.v;
        *(bf16x16*)(Pl + dst) = lo.v;
    }
}

// ---------------------------------------------------------------------------
// Stage 2: exact fp32 squared patch norms.
//   blockIdx.y == 0 : Rq[j] = ||q_j||^2 (source)   -- added per COLUMN
//   blockIdx.y == 1 : Rp[i] = ||p_i||^2 (target)   -- added per ROW
// ---------------------------------------------------------------------------
__global__ __launch_bounds__(256) void norms_kernel(
    const float* __restrict__ src, const float* __restrict__ tgt,
    float* __restrict__ Rq, float* __restrict__ Rp)
{
    const int i = blockIdx.x * 256 + threadIdx.x;
    if (i >= NPTS) return;
    const float* map = blockIdx.y ? tgt : src;
    float* out       = blockIdx.y ? Rp  : Rq;
    const int y = i >> 7, x = i & 127;
    float acc = 0.0f;
#pragma unroll
    for (int s = 0; s < 9; ++s) {
        const int off = clamp127(y + s / 3 - 1) * 128 + clamp127(x + s % 3 - 1);
#pragma unroll
        for (int ch = 0; ch < 32; ++ch) {
            const float v = map[ch * NPTS + off];
            acc = fmaf(v, v, acc);
        }
    }
    out[i] = acc;
}

// ---------------------------------------------------------------------------
// CDNA5 async global->LDS copy of one hi+lo P tile (18432 B, 1152 x b128),
// tracked by ASYNCcnt. LDS address = low 32 bits of the flat shared pointer
// (ISA 10.2: LDS aperture truncation).
// ---------------------------------------------------------------------------
__device__ __forceinline__ void async_stage_tile(
    const __bf16* __restrict__ Ph, const __bf16* __restrict__ Pl,
    int u, __bf16* buf, int tid)
{
    const char* gh = (const char*)(Ph + (size_t)u * (KSTEPS * BLK));
    const char* gl = (const char*)(Pl + (size_t)u * (KSTEPS * BLK));
    for (int idx = tid; idx < TILE_CHUNKS; idx += 256) {
        const char* g = (idx < TILE_CHUNKS / 2) ? (gh + idx * 16)
                                                : (gl + (idx - TILE_CHUNKS / 2) * 16);
        const unsigned lds_addr = (unsigned)(uintptr_t)((char*)buf + idx * 16);
        const unsigned long long gaddr = (unsigned long long)(uintptr_t)g;
        asm volatile("global_load_async_to_lds_b128 %0, %1, off"
                     :: "v"(lds_addr), "v"(gaddr) : "memory");
    }
}

__device__ __forceinline__ void wait_asynccnt0() {
    asm volatile("s_wait_asynccnt 0x0" ::: "memory");
}

// ---------------------------------------------------------------------------
// Stage 3: fused GEMM + argmin.
// 256 threads = 8 waves; wave w owns query row-tile t = blockIdx.x*8 + w and
// keeps its A fragments (Qh/Ql, 144 VGPRs) resident. P tiles are prefetched
// into a double-buffered LDS stage with async-to-LDS. Per tile: 36 DS reads
// feed 27 WMMAs on 3 independent accumulator chains; sched_group_barrier
// pins the skewed pipeline DS(24)-WMMA(9)-DS(12)-WMMA(9)-WMMA(9) so every
// WMMA burst runs on loads issued >=1 burst earlier (partial dscnt waits).
// score = Rq[j] - 2*S, running min/argmin in native C layout, shfl reduce.
// ---------------------------------------------------------------------------
__global__ __launch_bounds__(256, 1) void nnf_gemm_kernel(
    const __bf16* __restrict__ Qh, const __bf16* __restrict__ Ql,
    const __bf16* __restrict__ Ph, const __bf16* __restrict__ Pl,
    const float* __restrict__ Rq, const float* __restrict__ Rp,
    float* __restrict__ out)
{
    __shared__ __align__(32) __bf16 sB[2][2 * KSTEPS * BLK];  // double buffer

    const int tid  = threadIdx.x;
    const int wave = tid >> 5;
    const int lane = tid & 31;
    const int t    = blockIdx.x * 8 + wave;                // query row tile
    const int n    = lane & 15;                            // column slot

    // Resident A fragments for this wave's 16 query rows.
    bf16x16 ah[KSTEPS], al[KSTEPS];
    {
        const __bf16* qh = Qh + (size_t)t * (KSTEPS * BLK) + (size_t)lane * 16;
        const __bf16* ql = Ql + (size_t)t * (KSTEPS * BLK) + (size_t)lane * 16;
#pragma unroll
        for (int s = 0; s < KSTEPS; ++s) {
            ah[s] = *(const bf16x16*)(qh + s * BLK);
            al[s] = *(const bf16x16*)(ql + s * BLK);
        }
    }

    float minv[8];
    int   minj[8];
#pragma unroll
    for (int v = 0; v < 8; ++v) { minv[v] = 3.402823466e+38f; minj[v] = 0; }

    // Prime the pipeline: stage tile 0 into buffer 0.
    async_stage_tile(Ph, Pl, 0, sB[0], tid);

    for (int u = 0; u < NTILE; ++u) {
        wait_asynccnt0();      // this wave's async stores to LDS have landed
        __syncthreads();       // ...and everyone else's; prev reads all done

        if (u + 1 < NTILE)     // prefetch next tile into the other buffer
            async_stage_tile(Ph, Pl, u + 1, sB[(u + 1) & 1], tid);

        const __bf16* bbase = sB[u & 1];
        const float rq = Rq[u * 16 + n];   // source norm per COLUMN (the bug)

        // Batch-load all B fragments (36 ds_load_b128 in s order).
        bf16x16 bh[KSTEPS], bl[KSTEPS];
#pragma unroll
        for (int s = 0; s < KSTEPS; ++s) {
            bh[s] = *(const bf16x16*)(bbase + s * BLK + lane * 16);
            bl[s] = *(const bf16x16*)(bbase + KSTEPS * BLK + s * BLK + lane * 16);
        }

        // 27 WMMAs on 3 independent accumulator chains (s order matches loads).
        float8 c0, c1, c2;
#pragma unroll
        for (int v = 0; v < 8; ++v) { c0[v] = 0.0f; c1[v] = 0.0f; c2[v] = 0.0f; }
#pragma unroll
        for (int s = 0; s < KSTEPS; ++s) {
            c0 = __builtin_amdgcn_wmma_f32_16x16x32_bf16(false, ah[s], false, bh[s],
                                                         (short)0, c0, false, false);
            c1 = __builtin_amdgcn_wmma_f32_16x16x32_bf16(false, al[s], false, bh[s],
                                                         (short)0, c1, false, false);
            c2 = __builtin_amdgcn_wmma_f32_16x16x32_bf16(false, ah[s], false, bl[s],
                                                         (short)0, c2, false, false);
        }

        // Pin the skewed DS/WMMA pipeline for this iteration's region:
        // 24 DS reads (s0..5) | 9 WMMA (s0..2) | 12 DS reads (s6..8)
        // | 9 WMMA (s3..5) | 9 WMMA (s6..8).
        SGB(0x100, 24);
        SGB(0x008, 9);
        SGB(0x100, 12);
        SGB(0x008, 9);
        SGB(0x008, 9);

        const int jcol = u * 16 + n;
#pragma unroll
        for (int v = 0; v < 8; ++v) {
            const float score = fmaf(-2.0f, c0[v] + c1[v] + c2[v], rq);
            if (score < minv[v]) { minv[v] = score; minj[v] = jcol; }
        }
    }

    // Cross-lane (16-wide, within each half-wave) min+argmin reduction.
#pragma unroll
    for (int v = 0; v < 8; ++v) {
        float bv = minv[v];
        int   bj = minj[v];
#pragma unroll
        for (int off = 1; off < 16; off <<= 1) {
            const float ov = __shfl_xor(bv, off, 32);
            const int   oj = __shfl_xor(bj, off, 32);
            if (ov < bv || (ov == bv && oj < bj)) { bv = ov; bj = oj; }
        }
        if (n == 0) {
            const int m = v + ((lane >> 4) << 3);      // row within tile
            const int i = t * 16 + m;                  // query point index
            out[i]             = (float)(bj >> 7);     // nnf idy plane
            out[NPTS + i]      = (float)(bj & 127);    // nnf idx plane
            out[2 * NPTS + i]  = Rp[i] + bv;           // nnd (+ row-constant)
        }
    }
}

// ---------------------------------------------------------------------------
extern "C" void kernel_launch(void* const* d_in, const int* in_sizes, int n_in,
                              void* d_out, int out_size, void* d_ws, size_t ws_size,
                              hipStream_t stream)
{
    const float* src = (const float*)d_in[0];   // source_map (1,32,128,128) f32
    const float* tgt = (const float*)d_in[1];   // target_map (1,32,128,128) f32
    // d_in[2] = patch_size (== 3 per setup_inputs; kernels specialized for 3).
    (void)in_sizes; (void)n_in; (void)out_size; (void)ws_size;

    // Workspace layout (bytes): Qh | Ql | Ph | Pl | Rq | Rp  (~36.2 MB total)
    __bf16* Qh = (__bf16*)d_ws;
    __bf16* Ql = Qh + MATSZ;
    __bf16* Ph = Ql + MATSZ;
    __bf16* Pl = Ph + MATSZ;
    float*  Rq = (float*)(Pl + MATSZ);
    float*  Rp = Rq + NPTS;
    float*  out = (float*)d_out;

    dim3 g1((NTILE * KSTEPS * 32 + 255) / 256, 2);
    build_patches_kernel<<<g1, 256, 0, stream>>>(src, tgt, Qh, Ql, Ph, Pl);

    dim3 g2((NPTS + 255) / 256, 2);
    norms_kernel<<<g2, 256, 0, stream>>>(src, tgt, Rq, Rp);

    nnf_gemm_kernel<<<NTILE / 8, 256, 0, stream>>>(Qh, Ql, Ph, Pl, Rq, Rp, out);
}
